// RecurrentAutoEncoder_26353919328275
// MI455X (gfx1250) — matrix operand order repaired and checked by reading, not compiled, not run on hardware
//
#include <hip/hip_runtime.h>
#include <hip/hip_bf16.h>

// ---------------------------------------------------------------------------
// MI455X (gfx1250) seq2seq LSTM autoencoder.
//  - persistent workgroups: 16 batch rows each, full 256-step recurrence
//  - bf16 WMMA (v_wmma_f32_16x16x32_bf16), fp32 accumulate
//  - decoder dense layer folded into recurrent weights (one GEMM per step)
//  - c-state in VGPRs, h-state in LDS (bf16, WMMA A-fragment layout)
//  - weights pre-swizzled into per-wave-contiguous B fragments (1KB each)
//  - per-timestep opaque OFFSET (not pointer) so loads stay global_load_b128
//    (address-space inference intact) but cannot be hoisted+spilled
// ---------------------------------------------------------------------------

typedef __attribute__((ext_vector_type(16))) __bf16 v16bf;
typedef __attribute__((ext_vector_type(8)))  __bf16 v8bf;
typedef __attribute__((ext_vector_type(8)))  float  v8f;
typedef __attribute__((ext_vector_type(4)))  float  v4f;

#define B_   512
#define S_   128
#define F_   256
#define H_   512
#define G4H  2048            // 4*H

#define TM      16           // batch rows per workgroup
#define NWAVE   16           // waves per workgroup (wave32)
#define NTHREAD (NWAVE * 32)

#define HS (H_ + 8)          // LDS row stride (bf16 elems), 16B aligned rows
#define XS (F_ + 8)

// one 16(N)x32(K) bf16 fragment = 32 lanes * 16 elems
#define FRAG_ELEMS 512

// ------------------------------ prep kernels -------------------------------

__global__ void k_enc_bias(const float* __restrict__ bih,
                           const float* __restrict__ bhh,
                           float* __restrict__ out) {
    int i = blockIdx.x * blockDim.x + threadIdx.x;
    if (i < G4H) out[i] = bih[i] + bhh[i];
}

// WsumT[n][k] = dec_Whh[n][k] + sum_f dec_Wih[n][f] * dense_W[f][k]   (fp32 tmp)
// bias2[n]   = dec_bih[n] + dec_bhh[n] + sum_f dec_Wih[n][f] * dense_b[f]
__global__ void k_dec_fold(const float* __restrict__ dec_Wih,
                           const float* __restrict__ dec_Whh,
                           const float* __restrict__ dense_W,
                           const float* __restrict__ dense_b,
                           const float* __restrict__ dec_bih,
                           const float* __restrict__ dec_bhh,
                           float* __restrict__ WsumT,
                           float* __restrict__ bias2) {
    int idx = blockIdx.x * blockDim.x + threadIdx.x;
    if (idx >= G4H * H_) return;
    int n = idx / H_, k = idx % H_;
    float acc = dec_Whh[idx];
    for (int f = 0; f < F_; ++f)
        acc += dec_Wih[n * F_ + f] * dense_W[f * H_ + k];
    WsumT[idx] = acc;
    if (k == 0) {
        float b = dec_bih[n] + dec_bhh[n];
        for (int f = 0; f < F_; ++f)
            b += dec_Wih[n * F_ + f] * dense_b[f];
        bias2[n] = b;
    }
}

// Pack row-major fp32 W[N][K] into WMMA B-fragment order (bf16):
//   fragment(nt, kc): lane l supplies col = nt*16 + (l&15),
//                     k = kc*32 + (l>>4)*16 .. +15  (16 contiguous elems)
//   out[((nt*(K/32) + kc)*32 + l)*16 + i]
// -> each fragment is a contiguous 1KB block; lane l reads bytes [32l,32l+32).
__global__ void k_pack_b(const float* __restrict__ W,
                         __bf16* __restrict__ out, int N, int K) {
    int idx = blockIdx.x * blockDim.x + threadIdx.x;    // (N/16)*(K/32)*32
    int total = (N / 16) * (K / 32) * 32;
    if (idx >= total) return;
    int lane = idx & 31;
    int frag = idx >> 5;
    int KC = K / 32;
    int kc = frag % KC;
    int nt = frag / KC;
    int col = nt * 16 + (lane & 15);
    int kb  = kc * 32 + (lane >> 4) * 16;
    const float* src = W + (size_t)col * K + kb;
    __bf16* dst = out + (size_t)idx * 16;
#pragma unroll
    for (int i = 0; i < 16; ++i) dst[i] = (__bf16)src[i];
}

// ------------------------------ device helpers -----------------------------

__device__ __forceinline__ float fsig(float x) {
    return 1.0f / (1.0f + __expf(-x));
}
__device__ __forceinline__ float ftanh(float x) {
    float e = __expf(-2.0f * x);
    return (1.0f - e) / (1.0f + e);
}

// opaque zero: defeats loop-invariance proofs without destroying the base
// pointer's address space (loads stay global_load_*, only LOADcnt)
__device__ __forceinline__ int opq0() {
    int v = 0;
    asm volatile("" : "+s"(v));
    return v;
}

// A fragment (16x32 bf16) from LDS: lane<16 holds M=lane, K=k0..k0+15;
// lane>=16 holds M=lane-16, K+16. base pre-offset by lane.
__device__ __forceinline__ v16bf load_a_lds(const __bf16* base) {
    v8bf lo = *(const v8bf*)(base);
    v8bf hi = *(const v8bf*)(base + 8);
    v16bf r;
#pragma unroll
    for (int i = 0; i < 8; ++i) { r[i] = lo[i]; r[i + 8] = hi[i]; }
    return r;
}

// B fragment from packed stream: 32 contiguous bytes per lane
__device__ __forceinline__ v16bf load_b_pk(const __bf16* p) {
    v8bf lo = *(const v8bf*)(p);
    v8bf hi = *(const v8bf*)(p + 8);
    v16bf r;
#pragma unroll
    for (int i = 0; i < 8; ++i) { r[i] = lo[i]; r[i + 8] = hi[i]; }
    return r;
}

// ------------------------------ main kernel --------------------------------

__global__ void __launch_bounds__(NTHREAD, 1)
lstm_ae_kernel(const float*  __restrict__ x,        // [B][S][F] fp32
               const __bf16* __restrict__ Wih,      // packed frags, K=256
               const __bf16* __restrict__ Whh,      // packed frags, K=512
               const __bf16* __restrict__ Wsum,     // packed frags, K=512
               const __bf16* __restrict__ Wd,       // packed frags, K=512 (dense)
               const float*  __restrict__ enc_bias, // [2048]
               const float*  __restrict__ dec_bias, // [2048]
               const float*  __restrict__ dense_b,  // [256]
               float*        __restrict__ out)      // [B][S][F]
{
    __shared__ __align__(16) __bf16 h_tile[TM * HS];
    __shared__ __align__(16) __bf16 x_tile[TM * XS];

    const int tid  = threadIdx.x;
    const int wave = tid >> 5;           // 0..15; owns hidden cols [32w,32w+32)
    const int lane = tid & 31;
    const int l16  = lane & 15;
    const int hih  = lane >> 4;          // 0/1: lane half for WMMA layouts
    const int b0   = blockIdx.x * TM;

    for (int i = tid; i < TM * HS; i += NTHREAD) h_tile[i] = (__bf16)0.0f;
    __syncthreads();

    const int am  = l16;                 // A-fragment row
    const int akh = hih * 16;            // A-fragment K sub-offset
    const int lofs = lane * 16;          // lane offset inside a packed fragment

    // fragment elem offsets for this wave's 8 z-tiles (q = g*2+u)
    int fragIH[8], fragHH[8];
#pragma unroll
    for (int q = 0; q < 8; ++q) {
        int nt = (q >> 1) * 32 + wave * 2 + (q & 1);   // (g*H + w*32 + u*16)/16
        fragIH[q] = nt * (F_ / 32) * FRAG_ELEMS;       // K=256 stream
        fragHH[q] = nt * (H_ / 32) * FRAG_ELEMS;       // K=512 stream
    }
    const int fragD = wave * (H_ / 32) * FRAG_ELEMS;   // dense: nt = wave

    // pre-hoisted biases (loop-invariant, keep in VGPRs)
    float encB[8], decB[8];
#pragma unroll
    for (int q = 0; q < 8; ++q) {
        int n = (q >> 1) * H_ + wave * 32 + (q & 1) * 16 + l16;
        encB[q] = enc_bias[n];
        decB[q] = dec_bias[n];
    }
    const float dbv = dense_b[wave * 16 + l16];

    v8f c0, c1;                          // cell state, lives in VGPRs forever
#pragma unroll
    for (int e = 0; e < 8; ++e) { c0[e] = 0.0f; c1[e] = 0.0f; }

    // ================================ Encoder ================================
#pragma unroll 1
    for (int t = 0; t < S_; ++t) {
        // opaque per-step offsets: block hoisting, keep global address space
        const __bf16* pWih = Wih + opq0();
        const __bf16* pWhh = Whh + opq0();

        // stage x_t tile -> LDS bf16 (float4 global loads)
#pragma unroll
        for (int i = tid; i < TM * (F_ / 4); i += NTHREAD) {
            int m  = i / (F_ / 4);
            int f4 = (i % (F_ / 4)) * 4;
            v4f v = *(const v4f*)(x + ((size_t)(b0 + m) * S_ + t) * F_ + f4);
#pragma unroll
            for (int j = 0; j < 4; ++j) x_tile[m * XS + f4 + j] = (__bf16)v[j];
        }
        __syncthreads();   // x staged AND previous step's h writes visible

        v8f acc[8];        // tiles q = g*2+u : gate g in {i,f,g,o}, u in {0,1}
#pragma unroll
        for (int q = 0; q < 8; ++q)
#pragma unroll
            for (int e = 0; e < 8; ++e) acc[q][e] = 0.0f;

        // z += x_t @ Wih^T   (K = 256)
#pragma unroll 2
        for (int kc = 0; kc < F_ / 32; ++kc) {
            v16bf a = load_a_lds(&x_tile[am * XS + kc * 32 + akh]);
            v16bf b[8];
#pragma unroll
            for (int q = 0; q < 8; ++q)
                b[q] = load_b_pk(pWih + fragIH[q] + kc * FRAG_ELEMS + lofs);
#pragma unroll
            for (int q = 0; q < 8; ++q)
                acc[q] = __builtin_amdgcn_wmma_f32_16x16x32_bf16(
                    false, a, false, b[q], (short)0, acc[q], false, false);
        }
        // z += h @ Whh^T     (K = 512)
#pragma unroll 2
        for (int kc = 0; kc < H_ / 32; ++kc) {
            v16bf a = load_a_lds(&h_tile[am * HS + kc * 32 + akh]);
            v16bf b[8];
#pragma unroll
            for (int q = 0; q < 8; ++q)
                b[q] = load_b_pk(pWhh + fragHH[q] + kc * FRAG_ELEMS + lofs);
#pragma unroll
            for (int q = 0; q < 8; ++q)
                acc[q] = __builtin_amdgcn_wmma_f32_16x16x32_bf16(
                    false, a, false, b[q], (short)0, acc[q], false, false);
        }

        __syncthreads();   // everyone finished reading old h/x tiles

        // LSTM cell update; wave owns gates for its 32 hidden columns
#pragma unroll
        for (int u = 0; u < 2; ++u) {
            int col = wave * 32 + u * 16 + l16;
#pragma unroll
            for (int e = 0; e < 8; ++e) {
                float zi = acc[0 + u][e] + encB[0 + u];
                float zf = acc[2 + u][e] + encB[2 + u];
                float zg = acc[4 + u][e] + encB[4 + u];
                float zo = acc[6 + u][e] + encB[6 + u];
                float cp = (u == 0) ? c0[e] : c1[e];
                float cn = fsig(zf) * cp + fsig(zi) * ftanh(zg);
                if (u == 0) c0[e] = cn; else c1[e] = cn;
                float hn = fsig(zo) * ftanh(cn);
                int m = e + hih * 8;
                h_tile[m * HS + col] = (__bf16)hn;
            }
        }
    }

    __syncthreads();       // h_n fully written before decoder reads

    // ================================ Decoder ================================
#pragma unroll 1
    for (int t = 0; t < S_; ++t) {
        const __bf16* pWsum = Wsum + opq0();
        const __bf16* pWd   = Wd + opq0();

        v8f acc[8];
#pragma unroll
        for (int q = 0; q < 8; ++q)
#pragma unroll
            for (int e = 0; e < 8; ++e) acc[q][e] = 0.0f;

        // z = h @ (Wih·dW + Whh)^T   (K = 512)
#pragma unroll 2
        for (int kc = 0; kc < H_ / 32; ++kc) {
            v16bf a = load_a_lds(&h_tile[am * HS + kc * 32 + akh]);
            v16bf b[8];
#pragma unroll
            for (int q = 0; q < 8; ++q)
                b[q] = load_b_pk(pWsum + fragHH[q] + kc * FRAG_ELEMS + lofs);
#pragma unroll
            for (int q = 0; q < 8; ++q)
                acc[q] = __builtin_amdgcn_wmma_f32_16x16x32_bf16(
                    false, a, false, b[q], (short)0, acc[q], false, false);
        }

        __syncthreads();   // all waves done reading old h

#pragma unroll
        for (int u = 0; u < 2; ++u) {
            int col = wave * 32 + u * 16 + l16;
#pragma unroll
            for (int e = 0; e < 8; ++e) {
                float zi = acc[0 + u][e] + decB[0 + u];
                float zf = acc[2 + u][e] + decB[2 + u];
                float zg = acc[4 + u][e] + decB[4 + u];
                float zo = acc[6 + u][e] + decB[6 + u];
                float cp = (u == 0) ? c0[e] : c1[e];
                float cn = fsig(zf) * cp + fsig(zi) * ftanh(zg);
                if (u == 0) c0[e] = cn; else c1[e] = cn;
                float hn = fsig(zo) * ftanh(cn);
                int m = e + hih * 8;
                h_tile[m * HS + col] = (__bf16)hn;
            }
        }

        __syncthreads();   // h_new visible to all waves

        // emitted output: xi = h_new @ dense_W^T + dense_b; wave owns 16 cols
        v8f oacc;
#pragma unroll
        for (int e = 0; e < 8; ++e) oacc[e] = 0.0f;
#pragma unroll 2
        for (int kc = 0; kc < H_ / 32; ++kc) {
            v16bf a = load_a_lds(&h_tile[am * HS + kc * 32 + akh]);
            v16bf b = load_b_pk(pWd + fragD + kc * FRAG_ELEMS + lofs);
            oacc = __builtin_amdgcn_wmma_f32_16x16x32_bf16(
                false, a, false, b, (short)0, oacc, false, false);
        }
        {
            int col = wave * 16 + l16;
            int sout = S_ - 1 - t;   // torch.flip over time
#pragma unroll
            for (int e = 0; e < 8; ++e) {
                int m = e + hih * 8;
                out[((size_t)(b0 + m) * S_ + sout) * F_ + col] = oacc[e] + dbv;
            }
        }
        // next step's pre-write barrier covers the dense-GEMM readers of h.
    }
}

// ------------------------------ launch glue --------------------------------

extern "C" void kernel_launch(void* const* d_in, const int* in_sizes, int n_in,
                              void* d_out, int out_size, void* d_ws, size_t ws_size,
                              hipStream_t stream) {
    const float* x        = (const float*)d_in[0];
    const float* enc_Wih  = (const float*)d_in[1];
    const float* enc_Whh  = (const float*)d_in[2];
    const float* enc_bih  = (const float*)d_in[3];
    const float* enc_bhh  = (const float*)d_in[4];
    const float* dec_Wih  = (const float*)d_in[5];
    const float* dec_Whh  = (const float*)d_in[6];
    const float* dec_bih  = (const float*)d_in[7];
    const float* dec_bhh  = (const float*)d_in[8];
    const float* dense_W  = (const float*)d_in[9];
    const float* dense_b  = (const float*)d_in[10];
    float* out = (float*)d_out;

    // workspace layout (bytes)
    char* ws = (char*)d_ws;
    __bf16* pkWih   = (__bf16*)(ws);                          // 1 MB
    __bf16* pkWhh   = (__bf16*)(ws + (1u << 20));             // 2 MB
    __bf16* pkWsum  = (__bf16*)(ws + (3u << 20));             // 2 MB
    __bf16* pkWd    = (__bf16*)(ws + (5u << 20));             // 256 KB
    float*  wsEbias = (float*) (ws + (5u << 20) + (1u << 18));        // 8 KB
    float*  wsDbias = (float*) (ws + (5u << 20) + (1u << 18) + 8192); // 8 KB
    float*  wsWsumT = (float*) (ws + (6u << 20));             // 4 MB fp32 tmp

    // --- prep: bias merge, decoder dense fold, fragment packing ---
    k_enc_bias<<<(G4H + 255) / 256, 256, 0, stream>>>(enc_bih, enc_bhh, wsEbias);
    k_dec_fold<<<(G4H * H_ + 255) / 256, 256, 0, stream>>>(
        dec_Wih, dec_Whh, dense_W, dense_b, dec_bih, dec_bhh, wsWsumT, wsDbias);

    {   // pack enc_Wih: N=2048, K=256
        int n = (G4H / 16) * (F_ / 32) * 32;
        k_pack_b<<<(n + 255) / 256, 256, 0, stream>>>(enc_Wih, pkWih, G4H, F_);
    }
    {   // pack enc_Whh: N=2048, K=512
        int n = (G4H / 16) * (H_ / 32) * 32;
        k_pack_b<<<(n + 255) / 256, 256, 0, stream>>>(enc_Whh, pkWhh, G4H, H_);
    }
    {   // pack folded decoder weights: N=2048, K=512
        int n = (G4H / 16) * (H_ / 32) * 32;
        k_pack_b<<<(n + 255) / 256, 256, 0, stream>>>(wsWsumT, pkWsum, G4H, H_);
    }
    {   // pack dense_W: N=256, K=512
        int n = (F_ / 16) * (H_ / 32) * 32;
        k_pack_b<<<(n + 255) / 256, 256, 0, stream>>>(dense_W, pkWd, F_, H_);
    }

    // --- persistent recurrent kernel: 32 WGs x 16 waves, full 256 steps ---
    lstm_ae_kernel<<<B_ / TM, NTHREAD, 0, stream>>>(
        x, pkWih, pkWhh, pkWsum, pkWd, wsEbias, wsDbias, dense_b, out);
}
